// MultiHeadedSelfAttention_12214886990272
// MI455X (gfx1250) — matrix operand order repaired and checked by reading, not compiled, not run on hardware
//
#include <hip/hip_runtime.h>

#define B_  8
#define S_  1024
#define D_  768
#define H_  12
#define HD_ 64
#define NX_ (B_ * S_ * D_)   // 6,291,456 x elements
#define NW_ (D_ * D_)        // 589,824 per weight matrix

typedef _Float16 half_t;
typedef __attribute__((ext_vector_type(16))) _Float16 v16h;
typedef __attribute__((ext_vector_type(8)))  _Float16 v8h;
typedef __attribute__((ext_vector_type(8)))  float    v8f;

// ---------------------------------------------------------------------------
// gfx1250 async global->LDS copy (ASYNCcnt path). Inline asm per the bridge
// doc (portable across toolchains). VDST = per-lane LDS byte offset (low 32
// bits of a generic LDS pointer ARE the LDS offset per the aperture rules),
// VADDR = 64-bit global address, 16 bytes per lane.
// ---------------------------------------------------------------------------
__device__ __forceinline__ void async_copy_b128(unsigned lds_off, const void* g) {
    asm volatile("global_load_async_to_lds_b128 %0, %1, off"
                 :: "v"(lds_off), "v"(g)
                 : "memory");
}

__device__ __forceinline__ void wait_async0() {
#if __has_builtin(__builtin_amdgcn_s_wait_asynccnt)
    __builtin_amdgcn_s_wait_asynccnt(0);
#else
    asm volatile("s_wait_asynccnt 0" ::: "memory");
#endif
}

// ---------------------------------------------------------------------------
// Kernel 0: one-time f32 -> f16 conversion of x and Wq/Wk/Wv into workspace,
// so the GEMM hot loops are pure b128 loads + WMMA (no v_cvt in steady state).
// ---------------------------------------------------------------------------
__global__ __launch_bounds__(256) void cvt_kernel(
    const float* __restrict__ x,  const float* __restrict__ Wq,
    const float* __restrict__ Wk, const float* __restrict__ Wv,
    half_t* __restrict__ xh, half_t* __restrict__ wh)
{
    const size_t e = ((size_t)blockIdx.x * 256 + threadIdx.x) * 8;
    const float* src;
    half_t* dst;
    if (e < (size_t)NX_) {
        src = x + e;  dst = xh + e;
    } else {
        const size_t e2 = e - NX_;
        const int    m  = (int)(e2 / NW_);
        const size_t o  = e2 % NW_;
        src = (m == 0 ? Wq : (m == 1 ? Wk : Wv)) + o;
        dst = wh + e2;
    }
    const float4 a = *(const float4*)(src);
    const float4 b = *(const float4*)(src + 4);
    v8h r;
    r[0] = (half_t)a.x; r[1] = (half_t)a.y; r[2] = (half_t)a.z; r[3] = (half_t)a.w;
    r[4] = (half_t)b.x; r[5] = (half_t)b.y; r[6] = (half_t)b.z; r[7] = (half_t)b.w;
    *(v8h*)dst = r;
}

// ---------------------------------------------------------------------------
// Kernel 1: fused QKV projection from f16 inputs (unchanged from round 2).
// Q stored pre-scaled by 0.125 (B,H,S,HD); K (B,H,S,HD); V transposed
// (B,H,HD,S).
// ---------------------------------------------------------------------------
__global__ __launch_bounds__(256) void qkv_proj_kernel(
    const half_t* __restrict__ xh, const half_t* __restrict__ wh,
    const float* __restrict__ bq, const float* __restrict__ bk,
    const float* __restrict__ bv,
    half_t* __restrict__ qws, half_t* __restrict__ kws, half_t* __restrict__ vws)
{
    const int wave = threadIdx.x >> 5;
    const int lane = threadIdx.x & 31;
    const int wid  = blockIdx.x * 8 + wave;   // 18432 waves
    const int mt   = wid / 36;                // 512 M-tiles over (B*S)
    const int t    = wid % 36;
    const int mat  = t / 12;                  // 0=Q, 1=K, 2=V
    const int n64  = (t % 12) * 64;

    const half_t* W    = wh + (size_t)mat * NW_;
    const float*  bias = (mat == 0 ? bq : (mat == 1 ? bk : bv));

    const int arow = mt * 16 + (lane & 15);
    const int akb  = (lane >> 4) * 16;
    const int bcol = lane & 15;
    const int bk16 = (lane >> 4) * 16;

    v8f acc[4] = {};
    for (int k0 = 0; k0 < D_; k0 += 32) {
        const half_t* xp = xh + (size_t)arow * D_ + k0 + akb;
        const v8h lo = *(const v8h*)(xp);
        const v8h hi = *(const v8h*)(xp + 8);
        v16h a;
#pragma unroll
        for (int i = 0; i < 8; ++i) { a[i] = lo[i]; a[8 + i] = hi[i]; }
#pragma unroll
        for (int j = 0; j < 4; ++j) {
            const half_t* wp = W + (size_t)(n64 + j * 16 + bcol) * D_ + k0 + bk16;
            const v8h wl = *(const v8h*)(wp);
            const v8h wu = *(const v8h*)(wp + 8);
            v16h bm;
#pragma unroll
            for (int i = 0; i < 8; ++i) { bm[i] = wl[i]; bm[8 + i] = wu[i]; }
            acc[j] = __builtin_amdgcn_wmma_f32_16x16x32_f16(
                false, a, false, bm, (short)0, acc[j], false, false);
        }
    }

    const int roff = (lane >> 4) * 8;
#pragma unroll
    for (int j = 0; j < 4; ++j) {
#pragma unroll
        for (int r = 0; r < 8; ++r) {
            const int row = mt * 16 + roff + r;
            const int b   = row >> 10;
            const int s   = row & (S_ - 1);
            const int n   = n64 + j * 16 + bcol;
            const int h   = n >> 6;
            const int hd  = n & 63;
            const float v = acc[j][r] + bias[n];
            if (mat == 0) {
                qws[(((size_t)b * H_ + h) * S_ + s) * HD_ + hd] = (half_t)(v * 0.125f);
            } else if (mat == 1) {
                kws[(((size_t)b * H_ + h) * S_ + s) * HD_ + hd] = (half_t)v;
            } else {
                vws[(((size_t)b * H_ + h) * HD_ + hd) * S_ + s] = (half_t)v;
            }
        }
    }
}

// ---------------------------------------------------------------------------
// Kernel 2: cooperative attention. 4 waves / block, each wave owns a 32-query
// tile; the block shares each 32-key K/V tile via LDS, filled with
// double-buffered global_load_async_to_lds_b128 (ASYNCcnt) so the copy of
// stage i+1 overlaps the 16 WMMAs + softmax of stage i.
// Softmax without max shift (scores O(1); masked scores underflow to 0):
// lane-local row sums, one cross-lane reduction at the end.
// ---------------------------------------------------------------------------
__global__ __launch_bounds__(128) void attn_kernel(
    const half_t* __restrict__ qws, const half_t* __restrict__ kws,
    const half_t* __restrict__ vws, const int* __restrict__ mask,
    float* __restrict__ out)
{
    __shared__ half_t lds_k[2][32 * 64];   // [buf][key*64 + hd]   2 x 4KB
    __shared__ half_t lds_v[2][64 * 32];   // [buf][hd*32  + key]  2 x 4KB
    __shared__ half_t lds_p[4][32 * 32];   // per-wave P tiles     4 x 2KB

    const int w    = threadIdx.x >> 5;     // wave in block (0..3)
    const int lane = threadIdx.x & 31;
    const int qblk = blockIdx.x % (S_ / 128);
    const int h    = (blockIdx.x / (S_ / 128)) % H_;
    const int b    = blockIdx.x / ((S_ / 128) * H_);

    const size_t bh = (size_t)b * H_ + h;
    const int q0 = qblk * 128 + w * 32;                        // this wave's queries
    const half_t* qbase = qws + (bh * S_ + (size_t)q0) * HD_;
    const half_t* kbase = kws + bh * S_ * HD_;
    const half_t* vbase = vws + bh * HD_ * S_;

    const unsigned lk_off = (unsigned)(size_t)&lds_k[0][0];
    const unsigned lv_off = (unsigned)(size_t)&lds_v[0][0];

    const int arow = lane & 15;
    const int akb  = (lane >> 4) * 16;
    const int bcol = lane & 15;
    const int bk16 = (lane >> 4) * 16;
    const int roff = (lane >> 4) * 8;

    // ---- async stage issue: this wave's share of the 32-key K and V tiles
    auto issue_stage = [&](int kb, int buf) {
        // K tile: keys kb..kb+31 are one contiguous 4KB block of kws
        const char* ksrc = (const char*)kbase + (size_t)kb * (HD_ * 2);
        const unsigned kdst = lk_off + (unsigned)buf * (32 * 64 * 2);
#pragma unroll
        for (int i = 0; i < 2; ++i) {
            const unsigned off = (unsigned)((w * 2 + i) * 512 + lane * 16);
            async_copy_b128(kdst + off, ksrc + off);
        }
        // V tile: 64 rows (hd) x 64B of keys, global row stride S_*2 bytes
        const char* vsrc = (const char*)vbase + (size_t)kb * 2;
        const unsigned vdst = lv_off + (unsigned)buf * (64 * 32 * 2);
#pragma unroll
        for (int i = 0; i < 2; ++i) {
            const int row      = w * 16 + i * 8 + (lane >> 2);   // hd 0..63
            const unsigned c16 = (unsigned)(lane & 3) * 16;      // byte in row
            async_copy_b128(vdst + (unsigned)row * 64 + c16,
                            vsrc + (size_t)row * (S_ * 2) + c16);
        }
    };

    // ---- Q tiles in A-layout ----
    v16h qa[2][2];
#pragma unroll
    for (int qi = 0; qi < 2; ++qi) {
        const half_t* qp = qbase + (size_t)(qi * 16 + arow) * HD_ + akb;
        const v8h lo0 = *(const v8h*)(qp);
        const v8h hi0 = *(const v8h*)(qp + 8);
        const v8h lo1 = *(const v8h*)(qp + 32);
        const v8h hi1 = *(const v8h*)(qp + 40);
#pragma unroll
        for (int i = 0; i < 8; ++i) {
            qa[qi][0][i] = lo0[i]; qa[qi][0][8 + i] = hi0[i];
            qa[qi][1][i] = lo1[i]; qa[qi][1][8 + i] = hi1[i];
        }
    }

    float lsum[2][8];
#pragma unroll
    for (int qi = 0; qi < 2; ++qi)
#pragma unroll
        for (int r = 0; r < 8; ++r) lsum[qi][r] = 0.0f;
    v8f o[2][4] = {};

    // ---- prologue: fill buffer 0 ----
    issue_stage(0, 0);
    wait_async0();
    __syncthreads();

    int cur = 0;
    for (int kb = 0; kb < S_; kb += 32) {
        const bool more = (kb + 32 < S_);
        if (more) issue_stage(kb + 32, cur ^ 1);   // overlaps compute below

        const half_t* lk = &lds_k[cur][0];
        const half_t* lv = &lds_v[cur][0];

        // ---- K B-matrices from LDS (2 key tiles x 2 hd chunks) ----
        v16h kbm[4];
#pragma unroll
        for (int t = 0; t < 2; ++t) {
            const half_t* kp = lk + (size_t)(t * 16 + bcol) * 64 + bk16;
            const v8h a0 = *(const v8h*)(kp);      const v8h a1 = *(const v8h*)(kp + 8);
            const v8h a2 = *(const v8h*)(kp + 32); const v8h a3 = *(const v8h*)(kp + 40);
#pragma unroll
            for (int i = 0; i < 8; ++i) {
                kbm[t * 2 + 0][i] = a0[i]; kbm[t * 2 + 0][8 + i] = a1[i];
                kbm[t * 2 + 1][i] = a2[i]; kbm[t * 2 + 1][8 + i] = a3[i];
            }
        }

        // ---- scores: 4 tiles (2 query sub-tiles x 2 key tiles), K=64 each
        v8f s[2][2];
#pragma unroll
        for (int qi = 0; qi < 2; ++qi)
#pragma unroll
            for (int t = 0; t < 2; ++t) {
                v8f acc = {};
                acc = __builtin_amdgcn_wmma_f32_16x16x32_f16(
                    false, qa[qi][0], false, kbm[t * 2 + 0], (short)0, acc, false, false);
                acc = __builtin_amdgcn_wmma_f32_16x16x32_f16(
                    false, qa[qi][1], false, kbm[t * 2 + 1], (short)0, acc, false, false);
                s[qi][t] = acc;
            }

        // ---- mask, exp (no shift), lane-local row sums, P -> LDS ----
        const float pen0 = 10000.0f * (1.0f - (float)mask[b * S_ + kb + bcol]);
        const float pen1 = 10000.0f * (1.0f - (float)mask[b * S_ + kb + 16 + bcol]);
#pragma unroll
        for (int qi = 0; qi < 2; ++qi) {
#pragma unroll
            for (int r = 0; r < 8; ++r) {
                const float p0 = __expf(s[qi][0][r] - pen0);
                const float p1 = __expf(s[qi][1][r] - pen1);
                lsum[qi][r] += p0 + p1;
                lds_p[w][(qi * 16 + roff + r) * 32 + bcol]      = (half_t)p0;
                lds_p[w][(qi * 16 + roff + r) * 32 + bcol + 16] = (half_t)p1;
            }
        }

        // P is wave-private and DS ops are in-order within a wave: no barrier.
        v16h pa[2];
#pragma unroll
        for (int qi = 0; qi < 2; ++qi) {
            const half_t* pp = &lds_p[w][(qi * 16 + arow) * 32 + akb];
            const v8h lo = *(const v8h*)(pp);
            const v8h hi = *(const v8h*)(pp + 8);
#pragma unroll
            for (int i = 0; i < 8; ++i) { pa[qi][i] = lo[i]; pa[qi][8 + i] = hi[i]; }
        }

        // ---- O += P @ V from the shared LDS V tile ----
#pragma unroll
        for (int j = 0; j < 4; ++j) {
            const half_t* vp = lv + (size_t)(j * 16 + bcol) * 32 + bk16;
            const v8h v0 = *(const v8h*)(vp);
            const v8h v1 = *(const v8h*)(vp + 8);
            v16h vb;
#pragma unroll
            for (int i = 0; i < 8; ++i) { vb[i] = v0[i]; vb[8 + i] = v1[i]; }
#pragma unroll
            for (int qi = 0; qi < 2; ++qi)
                o[qi][j] = __builtin_amdgcn_wmma_f32_16x16x32_f16(
                    false, pa[qi], false, vb, (short)0, o[qi][j], false, false);
        }

        // ---- double-buffer handoff: own copies done, then block barrier ----
        if (more) wait_async0();
        __syncthreads();
        cur ^= 1;
    }

    // ---- one end-of-kernel row-sum reduction, normalize, store ----
#pragma unroll
    for (int qi = 0; qi < 2; ++qi) {
#pragma unroll
        for (int r = 0; r < 8; ++r) {
            float l = lsum[qi][r];
#pragma unroll
            for (int i = 1; i < 16; i <<= 1) l += __shfl_xor(l, i, 32);
            const float inv = 1.0f / l;
            const int srow = q0 + qi * 16 + roff + r;
#pragma unroll
            for (int j = 0; j < 4; ++j) {
                const int d = h * 64 + j * 16 + bcol;
                out[((size_t)b * S_ + srow) * D_ + d] = o[qi][j][r] * inv;
            }
        }
    }
}

extern "C" void kernel_launch(void* const* d_in, const int* in_sizes, int n_in,
                              void* d_out, int out_size, void* d_ws, size_t ws_size,
                              hipStream_t stream) {
    (void)in_sizes; (void)n_in; (void)out_size; (void)ws_size;
    const float* x   = (const float*)d_in[0];
    const int*  mask = (const int*) d_in[1];
    const float* Wq  = (const float*)d_in[2];
    const float* bq  = (const float*)d_in[3];
    const float* Wk  = (const float*)d_in[4];
    const float* bk  = (const float*)d_in[5];
    const float* Wv  = (const float*)d_in[6];
    const float* bv  = (const float*)d_in[7];
    float* out = (float*)d_out;

    const size_t QKV = (size_t)B_ * H_ * S_ * HD_;   // 6,291,456 f16 elems
    half_t* qws = (half_t*)d_ws;
    half_t* kws = qws + QKV;
    half_t* vws = kws + QKV;
    half_t* xh  = vws + QKV;                          // NX_ == QKV elems
    half_t* wh  = xh + NX_;                           // 3 * NW_ elems
    // total workspace: 4*QKV + 3*NW_ halves ~= 53.9 MB

    cvt_kernel<<<dim3(3936), dim3(256), 0, stream>>>(x, Wq, Wk, Wv, xh, wh);

    qkv_proj_kernel<<<dim3(2304), dim3(256), 0, stream>>>(
        xh, wh, bq, bk, bv, qws, kws, vws);

    // B*H*(S/128) = 768 blocks, 4 waves each (one 32-query tile per wave)
    attn_kernel<<<dim3(768), dim3(128), 0, stream>>>(qws, kws, vws, mask, out);
}
